// CaptionDecoder_695784701977
// MI455X (gfx1250) — matrix operand order repaired and checked by reading, not compiled, not run on hardware
//
#include <hip/hip_runtime.h>
#include <hip/hip_bf16.h>
#include <math.h>

#define VOCAB   30522
#define EMBED   512
#define HIDDEN  512
#define BATCH   32
#define TLEN    64
#define START_TOKEN 101
#define NBLOCKS 64
#define NTHREADS 256
#define NWAVES  (NBLOCKS * NTHREADS / 32)   // 512 waves
#define NTHREADS_TOTAL (NBLOCKS * NTHREADS) // 16384

static_assert(NTHREADS_TOTAL == BATCH * HIDDEN, "pointwise mapping");

typedef __bf16 bf16_t;
typedef bf16_t v16bf __attribute__((ext_vector_type(16)));
typedef bf16_t v8bf  __attribute__((ext_vector_type(8)));
typedef float  v8f   __attribute__((ext_vector_type(8)));

// ---------------- workspace layout (byte offsets) ----------------
#define OFF_CTRL      0u
#define OFF_AMAX      64u
#define OFF_TOKENS    512u
#define OFF_XBF       4096u
#define OFF_H0BF      (OFF_XBF  + 32768u)
#define OFF_H1BF      (OFF_H0BF + 32768u)
#define OFF_C0        (OFF_H1BF + 32768u)
#define OFF_C1        (OFF_C0 + 65536u)
#define OFF_G0        (OFF_C1 + 65536u)
#define OFF_G1        (OFF_G0 + 262144u)
// packed bf16 weights (fragment-major), 1 MiB aligned start
#define OFF_PK_WIH0   (1u << 20)
#define PK_LSTM_BYTES (2048u * 512u * 2u)            // 2 MiB each
#define OFF_PK_WHH0   (OFF_PK_WIH0 + PK_LSTM_BYTES)
#define OFF_PK_WIH1   (OFF_PK_WHH0 + PK_LSTM_BYTES)
#define OFF_PK_WHH1   (OFF_PK_WIH1 + PK_LSTM_BYTES)
#define OFF_PK_FCW    (OFF_PK_WHH1 + PK_LSTM_BYTES)
#define FC_NTILES     ((VOCAB + 15) / 16)            // 1908
#define PK_FC_BYTES   ((size_t)FC_NTILES * 8192u * 2u)
#define WS_NEEDED     ((size_t)OFF_PK_FCW + PK_FC_BYTES)

__device__ __forceinline__ float sigmoid_(float x) { return 1.0f / (1.0f + expf(-x)); }

// ---- device-wide split barrier (sense-reversing, counters in d_ws) ----
__device__ __forceinline__ void grid_sync(unsigned* cnt, unsigned* sense, unsigned& epoch) {
  __syncthreads();
  ++epoch;
  if (threadIdx.x == 0) {
    __threadfence();
    unsigned arrived = atomicAdd(cnt, 1u);
    if (arrived == NBLOCKS - 1) {
      atomicExch(cnt, 0u);
      __threadfence();
      atomicExch(sense, epoch);
    } else {
      while (atomicAdd(sense, 0u) < epoch) __builtin_amdgcn_s_sleep(1);
    }
  }
  __syncthreads();
}

// ---- WMMA fragment loaders (contiguous k order, same for A and B) ----
// A: 16x32 bf16 row-major tile; lane m = lane&15, half = lane>>4;
// element e holds K = half*16 + e  -> two contiguous 16B reads.
__device__ __forceinline__ v16bf load_a_frag(const bf16_t* A, int row0, int k0, int ld) {
  int lane = threadIdx.x & 31;
  int m  = row0 + (lane & 15);
  int hf = lane >> 4;
  const v8bf* p0 = reinterpret_cast<const v8bf*>(A + m * ld + k0 + hf * 16);
  const v8bf* p1 = reinterpret_cast<const v8bf*>(A + m * ld + k0 + hf * 16 + 8);
  v8bf c0 = *p0, c1 = *p1;
  return __builtin_shufflevector(c0, c1, 0,1,2,3,4,5,6,7,8,9,10,11,12,13,14,15);
}

// B (direct, fp32 source): B[k][n] = W[n0+n][k0+k]; lanes 0-15 K=0..15, 16-31 K=16..31.
__device__ __forceinline__ v16bf load_b_frag_cvt(const float* __restrict__ W,
                                                 int n0, int k0, int ld, int nmax) {
  int lane = threadIdx.x & 31;
  int n = n0 + (lane & 15);
  n = (n > nmax) ? nmax : n;
  int hf = lane >> 4;
  const float4* p = reinterpret_cast<const float4*>(W + (size_t)n * ld + k0 + hf * 16);
  float4 q0 = p[0], q1 = p[1], q2 = p[2], q3 = p[3];
  v16bf b;
  b[0]=(bf16_t)q0.x;  b[1]=(bf16_t)q0.y;  b[2]=(bf16_t)q0.z;  b[3]=(bf16_t)q0.w;
  b[4]=(bf16_t)q1.x;  b[5]=(bf16_t)q1.y;  b[6]=(bf16_t)q1.z;  b[7]=(bf16_t)q1.w;
  b[8]=(bf16_t)q2.x;  b[9]=(bf16_t)q2.y;  b[10]=(bf16_t)q2.z; b[11]=(bf16_t)q2.w;
  b[12]=(bf16_t)q3.x; b[13]=(bf16_t)q3.y; b[14]=(bf16_t)q3.z; b[15]=(bf16_t)q3.w;
  return b;
}

// B (packed, bf16 fragment-major): one coalesced 32B vector load per lane.
__device__ __forceinline__ v16bf load_b_frag_packed(const bf16_t* __restrict__ P,
                                                    int nt, int kc) {
  int lane = threadIdx.x & 31;
  const v16bf* p = reinterpret_cast<const v16bf*>(
      P + ((((size_t)nt * 16 + kc) * 32 + lane) << 4));
  return *p;
}

__device__ __forceinline__ v8f wmma_bf16(v16bf a, v16bf b, v8f c) {
  return __builtin_amdgcn_wmma_f32_16x16x32_bf16(
      /*neg_a=*/false, a, /*neg_b=*/false, b,
      /*c_mod=*/(short)0, c, /*reuse_a=*/false, /*reuse_b=*/false);
}

__device__ __forceinline__ unsigned long long shfl_xor_u64(unsigned long long v, int mask) {
  unsigned lo = (unsigned)v, hi = (unsigned)(v >> 32);
  lo = __shfl_xor(lo, mask, 32);
  hi = __shfl_xor(hi, mask, 32);
  return ((unsigned long long)hi << 32) | lo;
}

// pack (value, col) -> u64 so max() == (largest value, lowest col)  [jnp.argmax ties]
__device__ __forceinline__ unsigned long long pack_key(float v, int col) {
  unsigned u  = __float_as_uint(v);
  unsigned sv = (u >> 31) ? ~u : (u | 0x80000000u);
  return ((unsigned long long)sv << 32) | (unsigned)(0x7FFFFFFFu - col);
}

// fp32 [N][512] row-major -> bf16 fragment-major blocks of 8192 elems per 16-col tile.
// packed[((nt*16 + kc)*32 + lane)*16 + e] = bf16(W[nt*16 + (lane&15)][kc*32 + (lane>>4)*16 + e])
__device__ __forceinline__ void pack_weight(bf16_t* __restrict__ P,
                                            const float* __restrict__ W,
                                            int N, int gid) {
  int ntiles = (N + 15) >> 4;
  size_t total = (size_t)ntiles << 13;               // ntiles * 8192
  for (size_t p = gid; p < total; p += NTHREADS_TOTAL) {
    int e    = (int)(p & 15);
    int lane = (int)((p >> 4) & 31);
    int kc   = (int)((p >> 9) & 15);
    int nt   = (int)(p >> 13);
    int n = nt * 16 + (lane & 15);
    if (n > N - 1) n = N - 1;                        // clamp padded rows
    int k = kc * 32 + (lane >> 4) * 16 + e;
    P[p] = (bf16_t)W[(size_t)n * 512 + k];
  }
}

template <bool PACKED>
__global__ __launch_bounds__(NTHREADS)
void CaptionDecoder_persist(const float* __restrict__ fused,   // [32,512]
                            const int*   __restrict__ targets, // [32,64]
                            const int*   __restrict__ tf_mask, // [64]
                            const float* __restrict__ emb,     // [V,512]
                            const float* __restrict__ w_ih0, const float* __restrict__ w_hh0,
                            const float* __restrict__ b_ih0, const float* __restrict__ b_hh0,
                            const float* __restrict__ w_ih1, const float* __restrict__ w_hh1,
                            const float* __restrict__ b_ih1, const float* __restrict__ b_hh1,
                            const float* __restrict__ fc_w,  const float* __restrict__ fc_b,
                            float* __restrict__ out,           // [32,64,V]
                            char*  __restrict__ ws) {
  unsigned* cnt   = (unsigned*)(ws + OFF_CTRL);
  unsigned* sense = (unsigned*)(ws + OFF_CTRL + 4);
  unsigned long long* amax = (unsigned long long*)(ws + OFF_AMAX);
  int*    tokens  = (int*)   (ws + OFF_TOKENS);
  bf16_t* x_bf    = (bf16_t*)(ws + OFF_XBF);
  bf16_t* h0_bf   = (bf16_t*)(ws + OFF_H0BF);
  bf16_t* h1_bf   = (bf16_t*)(ws + OFF_H1BF);
  float*  c0      = (float*) (ws + OFF_C0);
  float*  c1      = (float*) (ws + OFF_C1);
  float*  gates0  = (float*) (ws + OFF_G0);
  float*  gates1  = (float*) (ws + OFF_G1);
  bf16_t* pk_ih0  = (bf16_t*)(ws + OFF_PK_WIH0);
  bf16_t* pk_hh0  = (bf16_t*)(ws + OFF_PK_WHH0);
  bf16_t* pk_ih1  = (bf16_t*)(ws + OFF_PK_WIH1);
  bf16_t* pk_hh1  = (bf16_t*)(ws + OFF_PK_WHH1);
  bf16_t* pk_fcw  = (bf16_t*)(ws + OFF_PK_FCW);

  const int tid = threadIdx.x;
  const int gid = blockIdx.x * NTHREADS + tid;   // 0..16383 == (b<<9)|k
  const int wid = gid >> 5;                      // global wave id 0..511
  unsigned epoch = 0;

  // ---- one-time prep: pack weights to bf16 fragment-major ----
  if constexpr (PACKED) {
    pack_weight(pk_ih0, w_ih0, 4 * HIDDEN, gid);
    pack_weight(pk_hh0, w_hh0, 4 * HIDDEN, gid);
    pack_weight(pk_ih1, w_ih1, 4 * HIDDEN, gid);
    pack_weight(pk_hh1, w_hh1, 4 * HIDDEN, gid);
    pack_weight(pk_fcw, fc_w, VOCAB, gid);
  }

  // ---- init: h0=h1=fused (bf16), c=0, tok=START ----
  {
    float hv = fused[gid];
    h0_bf[gid] = (bf16_t)hv;
    h1_bf[gid] = (bf16_t)hv;
    c0[gid] = 0.0f; c1[gid] = 0.0f;
    if (gid < BATCH) { tokens[gid] = START_TOKEN; amax[gid] = 0ull; }
  }
  grid_sync(cnt, sense, epoch);

  const int fc_tiles = 2 * FC_NTILES;   // 3816

  for (int t = 0; t < TLEN; ++t) {
    // ---- phase 0: embedding gather -> x_bf ----
    {
      int b = gid >> 9, k = gid & 511;
      int tok = tokens[b];
      x_bf[gid] = (bf16_t)emb[(size_t)tok * EMBED + k];
    }
    grid_sync(cnt, sense, epoch);

    // ---- phase 1: gates0 = x@w_ih0.T + h0@w_hh0.T + b0 ----
    if (wid < 256) {
      int mt = wid & 1, nt = wid >> 1;
      v8f acc = {};
      #pragma unroll 4
      for (int kc = 0; kc < 16; ++kc) {
        v16bf a  = load_a_frag(x_bf, mt * 16, kc * 32, EMBED);
        v16bf bf = PACKED ? load_b_frag_packed(pk_ih0, nt, kc)
                          : load_b_frag_cvt(w_ih0, nt * 16, kc * 32, EMBED, 4 * HIDDEN - 1);
        acc = wmma_bf16(a, bf, acc);
      }
      #pragma unroll 4
      for (int kc = 0; kc < 16; ++kc) {
        v16bf a  = load_a_frag(h0_bf, mt * 16, kc * 32, HIDDEN);
        v16bf bf = PACKED ? load_b_frag_packed(pk_hh0, nt, kc)
                          : load_b_frag_cvt(w_hh0, nt * 16, kc * 32, HIDDEN, 4 * HIDDEN - 1);
        acc = wmma_bf16(a, bf, acc);
      }
      int lane = tid & 31, n = lane & 15, hf = lane >> 4;
      int j = nt * 16 + n;
      float bias = b_ih0[j] + b_hh0[j];
      #pragma unroll
      for (int r = 0; r < 8; ++r)
        gates0[(mt * 16 + r + 8 * hf) * (4 * HIDDEN) + j] = acc[r] + bias;
    }
    grid_sync(cnt, sense, epoch);

    // ---- phase 2: LSTM cell-0 pointwise; reset argmax keys ----
    {
      int b = gid >> 9, k = gid & 511;
      const float* g = gates0 + b * (4 * HIDDEN);
      float ig = sigmoid_(g[k]);
      float fg = sigmoid_(g[512 + k]);
      float gg = tanhf(g[1024 + k]);
      float og = sigmoid_(g[1536 + k]);
      float c  = fg * c0[gid] + ig * gg;
      c0[gid] = c;
      h0_bf[gid] = (bf16_t)(og * tanhf(c));
      if (gid < BATCH) amax[gid] = 0ull;
    }
    grid_sync(cnt, sense, epoch);

    // ---- phase 3: gates1 = h0@w_ih1.T + h1@w_hh1.T + b1 ----
    if (wid < 256) {
      int mt = wid & 1, nt = wid >> 1;
      v8f acc = {};
      #pragma unroll 4
      for (int kc = 0; kc < 16; ++kc) {
        v16bf a  = load_a_frag(h0_bf, mt * 16, kc * 32, HIDDEN);
        v16bf bf = PACKED ? load_b_frag_packed(pk_ih1, nt, kc)
                          : load_b_frag_cvt(w_ih1, nt * 16, kc * 32, HIDDEN, 4 * HIDDEN - 1);
        acc = wmma_bf16(a, bf, acc);
      }
      #pragma unroll 4
      for (int kc = 0; kc < 16; ++kc) {
        v16bf a  = load_a_frag(h1_bf, mt * 16, kc * 32, HIDDEN);
        v16bf bf = PACKED ? load_b_frag_packed(pk_hh1, nt, kc)
                          : load_b_frag_cvt(w_hh1, nt * 16, kc * 32, HIDDEN, 4 * HIDDEN - 1);
        acc = wmma_bf16(a, bf, acc);
      }
      int lane = tid & 31, n = lane & 15, hf = lane >> 4;
      int j = nt * 16 + n;
      float bias = b_ih1[j] + b_hh1[j];
      #pragma unroll
      for (int r = 0; r < 8; ++r)
        gates1[(mt * 16 + r + 8 * hf) * (4 * HIDDEN) + j] = acc[r] + bias;
    }
    grid_sync(cnt, sense, epoch);

    // ---- phase 4: LSTM cell-1 pointwise ----
    {
      int b = gid >> 9, k = gid & 511;
      const float* g = gates1 + b * (4 * HIDDEN);
      float ig = sigmoid_(g[k]);
      float fg = sigmoid_(g[512 + k]);
      float gg = tanhf(g[1024 + k]);
      float og = sigmoid_(g[1536 + k]);
      float c  = fg * c1[gid] + ig * gg;
      c1[gid] = c;
      h1_bf[gid] = (bf16_t)(og * tanhf(c));
    }
    grid_sync(cnt, sense, epoch);

    // ---- phase 5: logits = h1@fc_w.T + fc_b ; store + fused argmax ----
    for (int tt = wid; tt < fc_tiles; tt += NWAVES) {
      int mt = tt & 1, nt = tt >> 1;
      v8f acc = {};
      #pragma unroll 4
      for (int kc = 0; kc < 16; ++kc) {
        v16bf a  = load_a_frag(h1_bf, mt * 16, kc * 32, HIDDEN);
        v16bf bf = PACKED ? load_b_frag_packed(pk_fcw, nt, kc)
                          : load_b_frag_cvt(fc_w, nt * 16, kc * 32, HIDDEN, VOCAB - 1);
        acc = wmma_bf16(a, bf, acc);
      }
      int lane = tid & 31, n = lane & 15, hf = lane >> 4;
      int col = nt * 16 + n;
      bool valid = col < VOCAB;
      float bias = fc_b[valid ? col : (VOCAB - 1)];
      #pragma unroll
      for (int r = 0; r < 8; ++r) {
        int m = mt * 16 + r + 8 * hf;               // batch row
        float v = valid ? (acc[r] + bias) : -INFINITY;
        if (valid) out[((size_t)m * TLEN + t) * VOCAB + col] = v;
        unsigned long long key = pack_key(v, valid ? col : 0x3FFFFFFF);
        #pragma unroll
        for (int msk = 1; msk <= 8; msk <<= 1) {
          unsigned long long ok = shfl_xor_u64(key, msk);
          if (ok > key) key = ok;
        }
        if (n == 0) atomicMax(&amax[m], key);
      }
    }
    grid_sync(cnt, sense, epoch);

    // ---- phase 6: next-token select (teacher forcing vs argmax) ----
    if (gid < BATCH) {
      unsigned long long key = amax[gid];
      int arg = (int)(0x7FFFFFFFu - (unsigned)(key & 0xFFFFFFFFull));
      bool tf = (tf_mask[t] > 0) && (t < TLEN - 1);
      int ti = (t + 1 < TLEN) ? (t + 1) : (TLEN - 1);
      int teach = targets[gid * TLEN + ti];
      tokens[gid] = tf ? teach : arg;
    }
    grid_sync(cnt, sense, epoch);
  }
}

extern "C" void kernel_launch(void* const* d_in, const int* in_sizes, int n_in,
                              void* d_out, int out_size, void* d_ws, size_t ws_size,
                              hipStream_t stream) {
  (void)in_sizes; (void)n_in; (void)out_size;
  hipMemsetAsync(d_ws, 0, 1024, stream);   // zero barrier/argmax/token control region
  const float* fused   = (const float*)d_in[0];
  const int*   targets = (const int*)d_in[1];
  const int*   tf_mask = (const int*)d_in[2];
  const float* emb     = (const float*)d_in[3];
  const float* w_ih0   = (const float*)d_in[4];
  const float* w_hh0   = (const float*)d_in[5];
  const float* b_ih0   = (const float*)d_in[6];
  const float* b_hh0   = (const float*)d_in[7];
  const float* w_ih1   = (const float*)d_in[8];
  const float* w_hh1   = (const float*)d_in[9];
  const float* b_ih1   = (const float*)d_in[10];
  const float* b_hh1   = (const float*)d_in[11];
  const float* fc_w    = (const float*)d_in[12];
  const float* fc_b    = (const float*)d_in[13];
  if (ws_size >= WS_NEEDED) {
    CaptionDecoder_persist<true><<<NBLOCKS, NTHREADS, 0, stream>>>(
        fused, targets, tf_mask, emb, w_ih0, w_hh0, b_ih0, b_hh0,
        w_ih1, w_hh1, b_ih1, b_hh1, fc_w, fc_b, (float*)d_out, (char*)d_ws);
  } else {
    CaptionDecoder_persist<false><<<NBLOCKS, NTHREADS, 0, stream>>>(
        fused, targets, tf_mask, emb, w_ih0, w_hh0, b_ih0, b_hh0,
        w_ih1, w_hh1, b_ih1, b_hh1, fc_w, fc_b, (float*)d_out, (char*)d_ws);
  }
}
